// MultiPeakAwareAttention_214748365225
// MI455X (gfx1250) — matrix operand order repaired and checked by reading, not compiled, not run on hardware
//
#include <hip/hip_runtime.h>
#include <stdint.h>

// ---------------- problem constants (from setup_inputs) ----------------
#define B_    4
#define N_    4096          // H*W
#define C_    128
#define H_    64
#define K_    8
#define ITERS_ 15
#define UVS   4104          // padded stride for u/v (N_+1 used)

typedef __bf16 bf16t;
typedef bf16t  bf16x16 __attribute__((ext_vector_type(16)));
typedef float  f32x8   __attribute__((ext_vector_type(8)));

__device__ __forceinline__ float clampf(float x, float lo, float hi) {
  return fminf(fmaxf(x, lo), hi);
}

__device__ __forceinline__ bf16t f2bf(float f) {
  union { float f; uint32_t u; } x; x.f = f;
  uint32_t r = x.u + 0x7fffu + ((x.u >> 16) & 1u);   // round-to-nearest-even
  unsigned short h = (unsigned short)(r >> 16);
  return __builtin_bit_cast(bf16t, h);
}

// online logsumexp accumulate
__device__ __forceinline__ void lse_add(float z, float& m, float& s) {
  if (z > m) { s = s * __expf(m - z) + 1.0f; m = z; }
  else       { s += __expf(z - m); }
}

// CDNA5 async global->LDS copy (16B) + wait  (CDNA5 ISA)
__device__ __forceinline__ void async_b128_to_lds(uint32_t lds_off, const void* gaddr) {
  asm volatile("global_load_async_to_lds_b128 %0, %1, off"
               :: "v"(lds_off), "v"((unsigned long long)(uintptr_t)gaddr)
               : "memory");
}
__device__ __forceinline__ void wait_async0() {
  asm volatile("s_wait_asynccnt 0" ::: "memory");
}

// ---------------- K1: l2-normalize -> bf16 ----------------
// one wave per feature row; 8 waves / block; rows = 2*B*N
__global__ __launch_bounds__(256) void k_normalize(const float* __restrict__ fa,
                                                   const float* __restrict__ fb,
                                                   bf16t* __restrict__ na,
                                                   bf16t* __restrict__ nb) {
  const int wave = threadIdx.x >> 5, lane = threadIdx.x & 31;
  long row = (long)blockIdx.x * 8 + wave;           // 0 .. 2*B*N-1
  const float* src; bf16t* dst; long r;
  if (row < (long)B_ * N_) { src = fa; dst = na; r = row; }
  else                     { src = fb; dst = nb; r = row - (long)B_ * N_; }
  const float4 vv = *(const float4*)(src + r * C_ + lane * 4);
  float ss = vv.x * vv.x + vv.y * vv.y + vv.z * vv.z + vv.w * vv.w;
  ss += __shfl_xor(ss, 16);
  ss += __shfl_xor(ss, 8);
  ss += __shfl_xor(ss, 4);
  ss += __shfl_xor(ss, 2);
  ss += __shfl_xor(ss, 1);
  const float inv = 1.0f / fmaxf(sqrtf(ss), 1e-12f);
  union { bf16t h[4]; uint2 u2; } o;
  o.h[0] = f2bf(vv.x * inv); o.h[1] = f2bf(vv.y * inv);
  o.h[2] = f2bf(vv.z * inv); o.h[3] = f2bf(vv.w * inv);
  *(uint2*)(dst + r * C_ + lane * 4) = o.u2;
}

// ---------------- K2: raw_sim = Â · B̂ᵀ  (bf16 WMMA, f32 accum) ----------------
// 128x128 tile / block (8 waves = 2M x 4N), K staged via async-to-LDS in 32 chunks
__global__ __launch_bounds__(256) void k_gemm_sim(const bf16t* __restrict__ A,
                                                  const bf16t* __restrict__ Bm,
                                                  float* __restrict__ sim) {
  __shared__ __align__(32) bf16t lA[128 * 32];   // [row][k]
  __shared__ __align__(32) bf16t lB[128 * 32];   // [n][k]  (B^T tile)
  const int b    = blockIdx.z;
  const int mblk = blockIdx.y * 128;
  const int nblk = blockIdx.x * 128;
  const int t    = threadIdx.x;
  const int lane = t & 31;
  const int wave = t >> 5;
  const int waveM = wave >> 2;          // 0..1  -> 64 rows each
  const int waveN = wave & 3;           // 0..3  -> 32 cols each

  f32x8 acc[4][2] = {};                 // 8 16x16 f32 tiles / wave

  const int lrow  = t >> 1;             // 0..127
  const int lpart = (t & 1) * 16;       // element offset 0 or 16
  const bf16t* gA0 = A  + ((size_t)b * N_ + (mblk + lrow)) * C_ + lpart;
  const bf16t* gB0 = Bm + ((size_t)b * N_ + (nblk + lrow)) * C_ + lpart;
  const uint32_t laA = (uint32_t)(uintptr_t)&lA[lrow * 32 + lpart];
  const uint32_t laB = (uint32_t)(uintptr_t)&lB[lrow * 32 + lpart];
  const int halfsel = (lane < 16) ? 0 : 1;

  for (int kc = 0; kc < C_; kc += 32) {
    async_b128_to_lds(laA,      gA0 + kc);
    async_b128_to_lds(laA + 16, gA0 + kc + 8);
    async_b128_to_lds(laB,      gB0 + kc);
    async_b128_to_lds(laB + 16, gB0 + kc + 8);
    if (kc + 32 < C_) {
      __builtin_prefetch(gA0 + kc + 32, 0, 0);   // global_prefetch next chunk
      __builtin_prefetch(gB0 + kc + 32, 0, 0);
    }
    wait_async0();
    __syncthreads();

    // B fragments: lane holds column n=lane%16; lanes<16: K0-15, lanes>=16: K16-31
    bf16x16 bfrag[2];
#pragma unroll
    for (int nt = 0; nt < 2; ++nt) {
      const int bn = waveN * 32 + nt * 16 + (lane & 15);
      bfrag[nt] = *(const bf16x16*)&lB[bn * 32 + halfsel * 16];
    }
#pragma unroll
    for (int mt = 0; mt < 4; ++mt) {
      // A fragment: lane holds row m=lane%16; lanes<16: K0-15, lanes>=16: K16-31
      const int ar = waveM * 64 + mt * 16 + (lane & 15);
      union { uint4 q[2]; bf16x16 v; } af;
      af.q[0] = *(const uint4*)&lA[ar * 32 + halfsel * 16];
      af.q[1] = *(const uint4*)&lA[ar * 32 + halfsel * 16 + 8];
#pragma unroll
      for (int nt = 0; nt < 2; ++nt) {
        acc[mt][nt] = __builtin_amdgcn_wmma_f32_16x16x32_bf16(
            false, af.v, false, bfrag[nt], (short)0, acc[mt][nt], false, false);
      }
    }
    __syncthreads();
  }

  // C/D layout: VGPR r -> M = r (lanes 0-15) / r+8 (lanes 16-31), N = lane%16
#pragma unroll
  for (int mt = 0; mt < 4; ++mt) {
#pragma unroll
    for (int nt = 0; nt < 2; ++nt) {
      const int ncol = nblk + waveN * 32 + nt * 16 + (lane & 15);
#pragma unroll
      for (int r = 0; r < 8; ++r) {
        const int mrow = mblk + waveM * 64 + mt * 16 + r + ((lane >= 16) ? 8 : 0);
        sim[((size_t)b * N_ + mrow) * (size_t)N_ + ncol] = acc[mt][nt][r];
      }
    }
  }
}

// ---------------- zero init ----------------
__global__ void k_zero(float* p, int n) {
  const int g = blockIdx.x * 256 + threadIdx.x;
  if (g < n) p[g] = 0.0f;
}

// block pair-reduction of (max, scaled-sum)
__device__ __forceinline__ void block_lse_reduce(float* sm, float* ss, int t,
                                                 float m, float s) {
  sm[t] = m; ss[t] = s; __syncthreads();
  for (int o = 128; o > 0; o >>= 1) {
    if (t < o) {
      const float m2 = sm[t + o], s2 = ss[t + o];
      const float M = fmaxf(sm[t], m2);
      float S;
      if (M == -__builtin_inff()) S = 0.0f;
      else S = ss[t] * __expf(sm[t] - M) + s2 * __expf(m2 - M);
      sm[t] = M; ss[t] = S;
    }
    __syncthreads();
  }
}

// ---------------- K3: Sinkhorn row update (one block per row i of batch b) --------
__global__ __launch_bounds__(256) void k_sink_row(const float* __restrict__ sim,
    const float* __restrict__ v, float* __restrict__ u,
    const float* __restrict__ dustbin, const float* __restrict__ temp, int b) {
  const int i = blockIdx.x;                 // 0..N_  (N_ = dustbin row)
  const int t = threadIdx.x;
  __shared__ float sm[256], ss[256];
  const float traw = *temp;
  const float tcl  = clampf(traw, 0.03f, 10.0f);
  const float ds   = clampf(*dustbin / tcl, -50.f, 50.f);
  const float* vb  = v + (size_t)b * UVS;
  float m = -__builtin_inff(), s = 0.0f;
  if (i < N_) {
    const float* row = sim + ((size_t)b * N_ + i) * (size_t)N_;
    for (int j = t; j < N_; j += 256) {
      float z = row[j] / traw;
      if (z != z) z = 0.0f;
      z = clampf(z, -50.f, 50.f) + vb[j];
      lse_add(z, m, s);
    }
    if (t == 0) lse_add(ds + vb[N_], m, s);
  } else {
    for (int j = t; j < N_; j += 256) lse_add(ds + vb[j], m, s);
    if (t == 0) lse_add(vb[N_], m, s);
  }
  block_lse_reduce(sm, ss, t, m, s);
  if (t == 0) {
    const float norm = -logf((float)(2 * N_));
    const float lmu  = (i < N_) ? norm : (logf((float)N_) + norm);
    u[(size_t)b * UVS + i] = lmu - (sm[0] + logf(ss[0]));
  }
}

// ---------------- K4a: Sinkhorn col partial (grid 16 x 8; 256 cols x 512 rows) ----
__global__ __launch_bounds__(256) void k_sink_col_part(const float* __restrict__ sim,
    const float* __restrict__ u, float* __restrict__ pm, float* __restrict__ ps,
    const float* __restrict__ temp, int b) {
  const int ic = blockIdx.y;                // row chunk (512 rows)
  const int t  = threadIdx.x;
  const int j  = blockIdx.x * 256 + t;      // owned column
  const float traw = *temp;
  const float* ub  = u + (size_t)b * UVS;
  __shared__ float lu[256];
  float m = -__builtin_inff(), s = 0.0f;
  for (int ib = ic * 512; ib < ic * 512 + 512; ib += 256) {
    __syncthreads();
    lu[t] = ub[ib + t];
    __syncthreads();
    const float* base = sim + ((size_t)b * N_ + ib) * (size_t)N_ + j;
    for (int ii = 0; ii < 256; ++ii) {
      float z = base[(size_t)ii * N_] / traw;   // coalesced across the block
      if (z != z) z = 0.0f;
      z = clampf(z, -50.f, 50.f) + lu[ii];
      lse_add(z, m, s);
    }
  }
  const size_t o = ((size_t)b * 8 + ic) * N_ + j;
  pm[o] = m; ps[o] = s;
}

// ---------------- K4b: merge col partials + dustbin row -> v[j] -------------------
__global__ __launch_bounds__(256) void k_sink_col_merge(const float* __restrict__ pm,
    const float* __restrict__ ps, const float* __restrict__ u, float* __restrict__ v,
    const float* __restrict__ dustbin, const float* __restrict__ temp, int b) {
  const int t = threadIdx.x;
  const int j = blockIdx.x * 256 + t;
  const float traw = *temp;
  const float tcl  = clampf(traw, 0.03f, 10.0f);
  const float ds   = clampf(*dustbin / tcl, -50.f, 50.f);
  const float* ub  = u + (size_t)b * UVS;
  float m = -__builtin_inff(), s = 0.0f;
#pragma unroll
  for (int ic = 0; ic < 8; ++ic) {
    const size_t o = ((size_t)b * 8 + ic) * N_ + j;
    const float m2 = pm[o], s2 = ps[o];
    const float M = fmaxf(m, m2);
    s = (M == -__builtin_inff()) ? 0.0f : s * __expf(m - M) + s2 * __expf(m2 - M);
    m = M;
  }
  lse_add(ds + ub[N_], m, s);                 // dustbin row i = N_
  v[(size_t)b * UVS + j] = -logf((float)(2 * N_)) - (m + logf(s));
}

// ---------------- K4c: dustbin column j = N_ --------------------------------------
__global__ __launch_bounds__(256) void k_sink_col_dust(const float* __restrict__ u,
    float* __restrict__ v, const float* __restrict__ dustbin,
    const float* __restrict__ temp, int b) {
  const int t = threadIdx.x;
  __shared__ float sm[256], ss[256];
  const float traw = *temp;
  const float tcl  = clampf(traw, 0.03f, 10.0f);
  const float ds   = clampf(*dustbin / tcl, -50.f, 50.f);
  const float* ub  = u + (size_t)b * UVS;
  float m = -__builtin_inff(), s = 0.0f;
  for (int i = t; i < N_; i += 256) lse_add(ds + ub[i], m, s);
  if (t == 0) lse_add(ub[N_], m, s);          // corner Z = 0
  block_lse_reduce(sm, ss, t, m, s);
  if (t == 0) {
    const float norm = -logf((float)(2 * N_));
    v[(size_t)b * UVS + N_] = (logf((float)N_) + norm) - (sm[0] + logf(ss[0]));
  }
}

// ---------------- K5: per-row finalize: entropy + top-8 ---------------------------
__global__ __launch_bounds__(256) void k_finalize(const float* __restrict__ sim,
    const float* __restrict__ u, const float* __restrict__ v,
    const float* __restrict__ temp, float* __restrict__ out_entropy,
    int* __restrict__ tki, float* __restrict__ tkl) {
  const int b = blockIdx.y, i = blockIdx.x, t = threadIdx.x;
  const float traw = *temp;
  const float ui = u[(size_t)b * UVS + i];
  const float* vb = v + (size_t)b * UVS;
  const float* row = sim + ((size_t)b * N_ + i) * (size_t)N_;
  float la[16];
  float psum = 0.0f;
#pragma unroll
  for (int jj = 0; jj < 16; ++jj) {
    const int j = jj * 256 + t;
    float z = row[j] / traw;
    if (z != z) z = 0.0f;
    z = clampf(z, -50.f, 50.f) + ui + vb[j];
    z = clampf(z, -50.f, 50.f);
    la[jj] = z;
    psum += __expf(fminf(z, 0.0f));
  }
  __shared__ float red[256];
  red[t] = psum; __syncthreads();
  for (int o = 128; o > 0; o >>= 1) { if (t < o) red[t] += red[t + o]; __syncthreads(); }
  const float rm = fmaxf(red[0], 1e-8f);
  __syncthreads();
  float ent = 0.0f;
#pragma unroll
  for (int jj = 0; jj < 16; ++jj) {
    const float p = __expf(fminf(la[jj], 0.0f)) / rm;
    ent -= p * logf(fmaxf(p, 1e-8f));
  }
  red[t] = ent; __syncthreads();
  for (int o = 128; o > 0; o >>= 1) { if (t < o) red[t] += red[t + o]; __syncthreads(); }
  if (t == 0) {
    const float valid = clampf(rm * (float)(2 * N_), 0.0f, 1.0f);
    out_entropy[(size_t)b * N_ + i] = red[0] * valid;
  }
  __syncthreads();
  // top-8 of log_assign (exp is monotone -> same index set as top-8 of ot_prob)
  __shared__ float smx[256];
  __shared__ int   sidx[256];
  unsigned used = 0;
  for (int k = 0; k < K_; ++k) {
    float bm = -1e30f; int bi = -1;
#pragma unroll
    for (int jj = 0; jj < 16; ++jj) {
      if (!((used >> jj) & 1u) && la[jj] > bm) { bm = la[jj]; bi = jj * 256 + t; }
    }
    smx[t] = bm; sidx[t] = bi; __syncthreads();
    for (int o = 128; o > 0; o >>= 1) {
      if (t < o && smx[t + o] > smx[t]) { smx[t] = smx[t + o]; sidx[t] = sidx[t + o]; }
      __syncthreads();
    }
    const float wv = smx[0]; const int wi = sidx[0];
    __syncthreads();
    if (t == 0) {
      const size_t o8 = ((size_t)b * N_ + i) * K_ + k;
      tkl[o8] = wv; tki[o8] = wi;
    }
    if (wi >= 0 && (wi & 255) == t) used |= (1u << (wi >> 8));
  }
}

// ---------------- K6: local geometric validation (7x7 box, no-pad counting) -------
__global__ __launch_bounds__(256) void k_geo(const float* __restrict__ posA,
    const float* __restrict__ posB, const int* __restrict__ tki,
    float* __restrict__ out_geo) {
  const int b = blockIdx.x >> 3;
  const int k = blockIdx.x & 7;
  const int t = threadIdx.x;
  __shared__ float dx[N_], dy[N_];           // 32 KB displacement field
  for (int n = t; n < N_; n += 256) {
    const int idx = tki[((size_t)b * N_ + n) * K_ + k];
    const float cx = posB[((size_t)b * N_ + idx) * 2 + 0];
    const float cy = posB[((size_t)b * N_ + idx) * 2 + 1];
    dx[n] = cx - posA[((size_t)b * N_ + n) * 2 + 0];
    dy[n] = cy - posA[((size_t)b * N_ + n) * 2 + 1];
  }
  __syncthreads();
  for (int n = t; n < N_; n += 256) {
    const int h = n >> 6, w = n & 63;
    const int h0 = (h > 3) ? h - 3 : 0, h1 = (h < 60) ? h + 3 : 63;
    const int w0 = (w > 3) ? w - 3 : 0, w1 = (w < 60) ? w + 3 : 63;
    float sx = 0, sy = 0, sxx = 0, syy = 0;
    for (int hh = h0; hh <= h1; ++hh)
      for (int ww = w0; ww <= w1; ++ww) {
        const float ax = dx[hh * 64 + ww], ay = dy[hh * 64 + ww];
        sx += ax; sxx += ax * ax; sy += ay; syy += ay * ay;
      }
    const float inv = 1.0f / (float)((h1 - h0 + 1) * (w1 - w0 + 1));
    const float mx = sx * inv, my = sy * inv;
    const float vx = fmaxf(sxx * inv - mx * mx, 0.0f);
    const float vy = fmaxf(syy * inv - my * my, 0.0f);
    out_geo[((size_t)b * N_ + n) * K_ + k] = 1.0f / (1.0f + (vx + vy) * 100.0f);
  }
}

// ---------------- K7: softmax-refined positions -----------------------------------
__global__ __launch_bounds__(256) void k_refine(const float* __restrict__ posB,
    const int* __restrict__ tki, const float* __restrict__ tkl,
    const float* __restrict__ geo, const float* __restrict__ geow,
    float* __restrict__ out_ref) {
  const int g = blockIdx.x * 256 + threadIdx.x;
  if (g >= B_ * N_) return;
  const int b = g >> 12;
  const float gw = clampf(*geow, 0.0f, 2.0f);
  float comb[K_]; float mx = -__builtin_inff();
#pragma unroll
  for (int k = 0; k < K_; ++k) {
    comb[k] = tkl[(size_t)g * K_ + k] + gw * geo[(size_t)g * K_ + k];
    mx = fmaxf(mx, comb[k]);
  }
  float se = 0.0f;
#pragma unroll
  for (int k = 0; k < K_; ++k) { comb[k] = __expf(comb[k] - mx); se += comb[k]; }
  float rx = 0.0f, ry = 0.0f;
#pragma unroll
  for (int k = 0; k < K_; ++k) {
    const int idx = tki[(size_t)g * K_ + k];
    const float wgt = comb[k] / se;
    rx += wgt * posB[((size_t)b * N_ + idx) * 2 + 0];
    ry += wgt * posB[((size_t)b * N_ + idx) * 2 + 1];
  }
  if (rx != rx) rx = 0.0f;
  if (ry != ry) ry = 0.0f;
  out_ref[(size_t)g * 2 + 0] = clampf(rx, -1.5f, 1.5f);
  out_ref[(size_t)g * 2 + 1] = clampf(ry, -1.5f, 1.5f);
}

// ---------------- host orchestration ----------------------------------------------
extern "C" void kernel_launch(void* const* d_in, const int* in_sizes, int n_in,
                              void* d_out, int out_size, void* d_ws, size_t ws_size,
                              hipStream_t stream) {
  (void)in_sizes; (void)n_in; (void)out_size; (void)ws_size;
  const float* feat_A  = (const float*)d_in[0];
  const float* feat_B  = (const float*)d_in[1];
  const float* pos_A   = (const float*)d_in[2];
  const float* pos_B   = (const float*)d_in[3];
  const float* dustbin = (const float*)d_in[4];
  const float* geo_w   = (const float*)d_in[5];
  const float* temp    = (const float*)d_in[6];

  float* out         = (float*)d_out;
  float* out_refined = out;                                      // [B,N,2]
  float* out_entropy = out + (size_t)B_ * N_ * 2;                // [B,N]
  float* out_sim     = out_entropy + (size_t)B_ * N_;            // [B,N,N]
  float* out_geo     = out_sim + (size_t)B_ * N_ * N_;           // [B,N,K]

  char* w = (char*)d_ws;
  bf16t* nA = (bf16t*)w;  w += (size_t)B_ * N_ * C_ * sizeof(bf16t);
  bf16t* nB = (bf16t*)w;  w += (size_t)B_ * N_ * C_ * sizeof(bf16t);
  float* uv = (float*)w;  w += (size_t)2 * B_ * UVS * sizeof(float);
  float* u  = uv;
  float* v  = uv + (size_t)B_ * UVS;
  int*   tki = (int*)w;   w += (size_t)B_ * N_ * K_ * sizeof(int);
  float* tkl = (float*)w; w += (size_t)B_ * N_ * K_ * sizeof(float);
  float* pm  = (float*)w; w += (size_t)B_ * 8 * N_ * sizeof(float);
  float* ps  = (float*)w;

  k_normalize<<<(2 * B_ * N_) / 8, 256, 0, stream>>>(feat_A, feat_B, nA, nB);
  k_gemm_sim<<<dim3(N_ / 128, N_ / 128, B_), 256, 0, stream>>>(nA, nB, out_sim);

  const int nuv = 2 * B_ * UVS;
  k_zero<<<(nuv + 255) / 256, 256, 0, stream>>>(uv, nuv);

  // batch-major Sinkhorn: keep each 67MB sim slice hot in the 192MB L2
  for (int b = 0; b < B_; ++b) {
    for (int it = 0; it < ITERS_; ++it) {
      k_sink_row<<<dim3(N_ + 1), 256, 0, stream>>>(out_sim, v, u, dustbin, temp, b);
      k_sink_col_part<<<dim3(16, 8), 256, 0, stream>>>(out_sim, u, pm, ps, temp, b);
      k_sink_col_merge<<<dim3(16), 256, 0, stream>>>(pm, ps, u, v, dustbin, temp, b);
      k_sink_col_dust<<<dim3(1), 256, 0, stream>>>(u, v, dustbin, temp, b);
    }
  }

  k_finalize<<<dim3(N_, B_), 256, 0, stream>>>(out_sim, u, v, temp, out_entropy, tki, tkl);
  k_geo<<<B_ * K_, 256, 0, stream>>>(pos_A, pos_B, tki, out_geo);
  k_refine<<<(B_ * N_ + 255) / 256, 256, 0, stream>>>(pos_B, tki, tkl, out_geo, geo_w,
                                                      out_refined);
}
